// RNN_74380243632463
// MI455X (gfx1250) — matrix operand order repaired and checked by reading, not compiled, not run on hardware
//
#include <hip/hip_runtime.h>
#include <cmath>

typedef __attribute__((ext_vector_type(16))) _Float16 v16h;
typedef __attribute__((ext_vector_type(8)))  _Float16 v8h;
typedef __attribute__((ext_vector_type(8)))  float    v8f;

constexpr int Bb = 64;    // batch
constexpr int Tt = 2048;  // time
constexpr int Ii = 256;   // input
constexpr int Hh = 256;   // hidden (layer 1)
constexpr int Oo = 128;   // output (layer 2)

// ---------------------------------------------------------------------------
// helpers
// ---------------------------------------------------------------------------
__device__ __forceinline__ v16h cat8(v8h lo, v8h hi) {
  v16h r;
#pragma unroll
  for (int i = 0; i < 8; ++i) { r[i] = lo[i]; r[i + 8] = hi[i]; }
  return r;
}

__device__ __forceinline__ v8f wmma16(v16h a, v16h b, v8f c) {
  // D = A(16x32 f16) * B(32x16 f16) + C(16x16 f32)
  return __builtin_amdgcn_wmma_f32_16x16x32_f16(
      /*neg_a=*/false, a, /*neg_b=*/false, b,
      /*c_mod=*/(short)0, c, /*reuse_a=*/false, /*reuse_b=*/false);
}

// Hardware tanh (gfx1250 V_TANH_F32) when the builtin exists; otherwise a
// short exp-based form — both far cheaper than libm tanhf on the serial path.
__device__ __forceinline__ float fast_tanh(float x) {
#if __has_builtin(__builtin_amdgcn_tanhf)
  return __builtin_amdgcn_tanhf(x);
#else
  float e = __expf(2.0f * x);
  return 1.0f - 2.0f / (e + 1.0f);
#endif
}

// ---------------------------------------------------------------------------
// f32 -> f16 weight conversion
// ---------------------------------------------------------------------------
__global__ void k_cvt(const float* __restrict__ in, _Float16* __restrict__ o, int n) {
  int i = blockIdx.x * 256 + threadIdx.x;
  if (i < n) o[i] = (_Float16)in[i];
}

// ---------------------------------------------------------------------------
// GEMM1: xp1[t*B+b][h] = x[b,t,:] @ W_ih1^T + (b_ih1 + b_hh1)
//   A rows r = t*B + b (f32 source, converted inline), B = W_ih1 (f16, [H][I])
//   one wave per 16x16 output tile, 8 waves per block
// ---------------------------------------------------------------------------
__global__ __launch_bounds__(256) void k_gemm_xp1(
    const float* __restrict__ x, const _Float16* __restrict__ w,
    const float* __restrict__ bi, const float* __restrict__ bh,
    float* __restrict__ out) {
  const int lane = threadIdx.x & 31;
  const int wv   = threadIdx.x >> 5;
  const int tile = blockIdx.x * 8 + wv;
  const int nt = tile & 15;          // 16 N-tiles over H=256
  const int rt = tile >> 4;
  const int n0 = nt * 16;
  const int r0 = rt * 16;
  const int col = lane & 15;
  const int g   = lane >> 4;

  // A row for this lane (two lanes share a row, different K halves)
  const int rowA = r0 + col;
  const int bidx = rowA & (Bb - 1);
  const int tidx = rowA >> 6;        // 64 = Bb
  const float* xrow = x + ((size_t)bidx * Tt + tidx) * Ii;

  v8f acc = {};
#pragma unroll
  for (int kk = 0; kk < Ii / 32; ++kk) {
    const int ka = kk * 32 + 8 * g;
    v16h a;
#pragma unroll
    for (int j = 0; j < 8; ++j) {
      a[j]     = (_Float16)xrow[ka + j];
      a[j + 8] = (_Float16)xrow[ka + 16 + j];
    }
    const _Float16* wr = w + (size_t)(n0 + col) * Ii + kk * 32 + 8 * g;
    v16h bm = cat8(*(const v8h*)wr, *(const v8h*)(wr + 16));
    acc = wmma16(a, bm, acc);
  }

  const float bias = bi[n0 + col] + bh[n0 + col];
  float* orow = out + (size_t)r0 * Hh + n0 + col;
#pragma unroll
  for (int i = 0; i < 8; ++i)
    orow[(size_t)(i + 8 * g) * Hh] = acc[i] + bias;
}

// ---------------------------------------------------------------------------
// GEMM2: xp2[t*B+b][o] = h1[t*B+b,:] @ W_ih2^T + (b_ih2 + b_hh2)
//   A rows are f16 (h1 produced by scan1)
// ---------------------------------------------------------------------------
__global__ __launch_bounds__(256) void k_gemm_xp2(
    const _Float16* __restrict__ h1, const _Float16* __restrict__ w,
    const float* __restrict__ bi, const float* __restrict__ bh,
    float* __restrict__ out) {
  const int lane = threadIdx.x & 31;
  const int wv   = threadIdx.x >> 5;
  const int tile = blockIdx.x * 8 + wv;
  const int nt = tile & 7;           // 8 N-tiles over O=128
  const int rt = tile >> 3;
  const int n0 = nt * 16;
  const int r0 = rt * 16;
  const int col = lane & 15;
  const int g   = lane >> 4;

  const _Float16* arow = h1 + (size_t)(r0 + col) * Hh;

  v8f acc = {};
#pragma unroll
  for (int kk = 0; kk < Hh / 32; ++kk) {
    const int ka = kk * 32 + 8 * g;
    v16h a = cat8(*(const v8h*)(arow + ka), *(const v8h*)(arow + ka + 16));
    const _Float16* wr = w + (size_t)(n0 + col) * Hh + kk * 32 + 8 * g;
    v16h bm = cat8(*(const v8h*)wr, *(const v8h*)(wr + 16));
    acc = wmma16(a, bm, acc);
  }

  const float bias = bi[n0 + col] + bh[n0 + col];
  float* orow = out + (size_t)r0 * Oo + n0 + col;
#pragma unroll
  for (int i = 0; i < 8; ++i)
    orow[(size_t)(i + 8 * g) * Oo] = acc[i] + bias;
}

// ---------------------------------------------------------------------------
// scan1: h_t = tanh(xp1_t + h_{t-1} @ W_hh1^T), 16 batch rows per workgroup.
//   16 waves = 16 N-tiles (H=256). W_hh1 fragments resident in VGPRs.
//   h state ping-pongs through padded LDS (f16); one barrier per step.
//   xp tile loads are software-pipelined 2 steps ahead + prefetched 8 ahead.
//   Emits h1 (f16) to global for GEMM2.
// ---------------------------------------------------------------------------
__global__ __launch_bounds__(512) void k_scan1(
    const float* __restrict__ xp, const _Float16* __restrict__ whh,
    _Float16* __restrict__ h1out) {
  constexpr int LP = Hh + 8;                         // 8-half pad -> bank skew
  __shared__ __align__(16) _Float16 hbuf[2][16][LP];

  const int lane = threadIdx.x & 31;
  const int wv   = threadIdx.x >> 5;                 // N-tile 0..15
  const int col  = lane & 15;
  const int g    = lane >> 4;
  const int n0   = wv * 16;
  const int b0   = blockIdx.x * 16;

  // Preload time-invariant B fragments (W_hh1^T tiles): 8 k-steps
  v16h bfrag[8];
#pragma unroll
  for (int kk = 0; kk < 8; ++kk) {
    const _Float16* wr = whh + (size_t)(n0 + col) * Hh + kk * 32 + 8 * g;
    bfrag[kk] = cat8(*(const v8h*)wr, *(const v8h*)(wr + 16));
  }

  // h0 = 0
  for (int i = threadIdx.x; i < 16 * LP; i += 512)
    ((_Float16*)hbuf[0])[i] = (_Float16)0.f;

  // Pipeline prologue: xp tiles for t=0, t=1
  v8f xb[2];
#pragma unroll
  for (int u = 0; u < 2; ++u) {
    const float* xr = xp + ((size_t)u * Bb + b0) * Hh + n0 + col;
#pragma unroll
    for (int i = 0; i < 8; ++i) xb[u][i] = xr[(size_t)(i + 8 * g) * Hh];
  }
  __syncthreads();

  for (int tb = 0; tb < Tt; tb += 2) {
#pragma unroll
    for (int u = 0; u < 2; ++u) {
      const int t = tb + u;

      // consume pipelined xp tile, then refill for t+2 (overlaps WMMA chain)
      v8f acc = xb[u];
      {
        const int tn = (t + 2 < Tt) ? t + 2 : t;
        const float* xr = xp + ((size_t)tn * Bb + b0) * Hh + n0 + col;
#pragma unroll
        for (int i = 0; i < 8; ++i) xb[u][i] = xr[(size_t)(i + 8 * g) * Hh];
      }
      // deep prefetch of the xp stream (one line per lane, covers wave tile)
      {
        const int tp = (t + 8 < Tt) ? t + 8 : t;
        __builtin_prefetch(xp + ((size_t)tp * Bb + b0 + col) * Hh + n0, 0, 3);
      }

      // h_{t-1} @ W_hh1^T : K = 256 -> 8 chained WMMAs (A from LDS buf[u])
#pragma unroll
      for (int kk = 0; kk < 8; ++kk) {
        const _Float16* ap = &hbuf[u][col][kk * 32 + 8 * g];
        v16h a = cat8(*(const v8h*)ap, *(const v8h*)(ap + 16));
        acc = wmma16(a, bfrag[kk], acc);
      }

      // tanh, publish new state to other buffer + h1 to global
      _Float16* hg = h1out + ((size_t)t * Bb + b0) * Hh + n0 + col;
#pragma unroll
      for (int i = 0; i < 8; ++i) {
        _Float16 hv = (_Float16)fast_tanh(acc[i]);
        hbuf[u ^ 1][i + 8 * g][n0 + col] = hv;
        hg[(size_t)(i + 8 * g) * Hh] = hv;
      }
      __syncthreads();
    }
  }
}

// ---------------------------------------------------------------------------
// scan2: o_t = tanh(xp2_t + o_{t-1} @ W_hh2^T), writes d_out[b*T+t][o] (f32)
//   8 waves = 8 N-tiles (O=128), K = 128 -> 4 WMMAs per step.
// ---------------------------------------------------------------------------
__global__ __launch_bounds__(256) void k_scan2(
    const float* __restrict__ xp, const _Float16* __restrict__ whh,
    float* __restrict__ out) {
  constexpr int LP = Oo + 8;
  __shared__ __align__(16) _Float16 obuf[2][16][LP];

  const int lane = threadIdx.x & 31;
  const int wv   = threadIdx.x >> 5;                 // N-tile 0..7
  const int col  = lane & 15;
  const int g    = lane >> 4;
  const int n0   = wv * 16;
  const int b0   = blockIdx.x * 16;

  v16h bfrag[4];
#pragma unroll
  for (int kk = 0; kk < 4; ++kk) {
    const _Float16* wr = whh + (size_t)(n0 + col) * Oo + kk * 32 + 8 * g;
    bfrag[kk] = cat8(*(const v8h*)wr, *(const v8h*)(wr + 16));
  }

  for (int i = threadIdx.x; i < 16 * LP; i += 256)
    ((_Float16*)obuf[0])[i] = (_Float16)0.f;

  v8f xb[2];
#pragma unroll
  for (int u = 0; u < 2; ++u) {
    const float* xr = xp + ((size_t)u * Bb + b0) * Oo + n0 + col;
#pragma unroll
    for (int i = 0; i < 8; ++i) xb[u][i] = xr[(size_t)(i + 8 * g) * Oo];
  }
  __syncthreads();

  for (int tb = 0; tb < Tt; tb += 2) {
#pragma unroll
    for (int u = 0; u < 2; ++u) {
      const int t = tb + u;

      v8f acc = xb[u];
      {
        const int tn = (t + 2 < Tt) ? t + 2 : t;
        const float* xr = xp + ((size_t)tn * Bb + b0) * Oo + n0 + col;
#pragma unroll
        for (int i = 0; i < 8; ++i) xb[u][i] = xr[(size_t)(i + 8 * g) * Oo];
      }
      {
        const int tp = (t + 8 < Tt) ? t + 8 : t;
        __builtin_prefetch(xp + ((size_t)tp * Bb + b0 + col) * Oo + n0, 0, 3);
      }

#pragma unroll
      for (int kk = 0; kk < 4; ++kk) {
        const _Float16* ap = &obuf[u][col][kk * 32 + 8 * g];
        v16h a = cat8(*(const v8h*)ap, *(const v8h*)(ap + 16));
        acc = wmma16(a, bfrag[kk], acc);
      }

#pragma unroll
      for (int i = 0; i < 8; ++i) {
        const float ov = fast_tanh(acc[i]);
        obuf[u ^ 1][i + 8 * g][n0 + col] = (_Float16)ov;
        // reference output order: out[b*T + t][o]
        out[((size_t)(b0 + i + 8 * g) * Tt + t) * Oo + n0 + col] = ov;
      }
      __syncthreads();
    }
  }
}

// ---------------------------------------------------------------------------
// launcher
// ---------------------------------------------------------------------------
extern "C" void kernel_launch(void* const* d_in, const int* in_sizes, int n_in,
                              void* d_out, int out_size, void* d_ws, size_t ws_size,
                              hipStream_t stream) {
  const float* x     = (const float*)d_in[0];
  const float* W_ih1 = (const float*)d_in[1];
  const float* W_hh1 = (const float*)d_in[2];
  const float* b_ih1 = (const float*)d_in[3];
  const float* b_hh1 = (const float*)d_in[4];
  const float* W_ih2 = (const float*)d_in[5];
  const float* W_hh2 = (const float*)d_in[6];
  const float* b_ih2 = (const float*)d_in[7];
  const float* b_hh2 = (const float*)d_in[8];
  float* out = (float*)d_out;

  char* ws = (char*)d_ws;
  size_t off = 0;
  float*    xp1 = (float*)(ws + off);    off += (size_t)Tt * Bb * Hh * 4;  // 128 MB
  float*    xp2 = (float*)(ws + off);    off += (size_t)Tt * Bb * Oo * 4;  //  64 MB
  _Float16* h1  = (_Float16*)(ws + off); off += (size_t)Tt * Bb * Hh * 2;  //  64 MB
  _Float16* w1i = (_Float16*)(ws + off); off += (size_t)Hh * Ii * 2;
  _Float16* w1h = (_Float16*)(ws + off); off += (size_t)Hh * Hh * 2;
  _Float16* w2i = (_Float16*)(ws + off); off += (size_t)Oo * Hh * 2;
  _Float16* w2h = (_Float16*)(ws + off); off += (size_t)Oo * Oo * 2;
  (void)ws_size; (void)in_sizes; (void)n_in; (void)out_size;

  // weights -> f16
  k_cvt<<<(Hh * Ii + 255) / 256, 256, 0, stream>>>(W_ih1, w1i, Hh * Ii);
  k_cvt<<<(Hh * Hh + 255) / 256, 256, 0, stream>>>(W_hh1, w1h, Hh * Hh);
  k_cvt<<<(Oo * Hh + 255) / 256, 256, 0, stream>>>(W_ih2, w2i, Oo * Hh);
  k_cvt<<<(Oo * Oo + 255) / 256, 256, 0, stream>>>(W_hh2, w2h, Oo * Oo);

  // parallel input projections
  k_gemm_xp1<<<(Tt * Bb / 16) * (Hh / 16) / 8, 256, 0, stream>>>(x, w1i, b_ih1, b_hh1, xp1);
  // layer-1 recurrence (4 WGPs, batch-split)
  k_scan1<<<Bb / 16, 512, 0, stream>>>(xp1, w1h, h1);
  // layer-2 input projection
  k_gemm_xp2<<<(Tt * Bb / 16) * (Oo / 16) / 8, 256, 0, stream>>>(h1, w2i, b_ih2, b_hh2, xp2);
  // layer-2 recurrence -> final output
  k_scan2<<<Bb / 16, 256, 0, stream>>>(xp2, w2h, out);
}
